// GemmaAttention_27616639713595
// MI455X (gfx1250) — matrix operand order repaired and checked by reading, not compiled, not run on hardware
//
#include <hip/hip_runtime.h>
#include <hip/hip_bf16.h>

// CDNA5 / gfx1250 — wave32, WMMA 16x16x32 bf16 (f32 accumulate), async
// global->LDS staging (ASYNCcnt) with double-buffered LDS pipelines and
// software-pipelined fragment loads.
// Pipeline: [GEMM1 hidden@Wqkv -> bf16 qkv] -> [RoPE+fold scaling/softcap]
//           -> [flash attention, windowed+causal, online softmax]
//           -> [GEMM2 O@Wo -> f32 out]

typedef __attribute__((ext_vector_type(16))) __bf16 v16bf;
typedef __attribute__((ext_vector_type(8)))  float  v8f;
typedef int v4i_ __attribute__((vector_size(16)));   // matches builtin proto

#define S_   2048
#define HID_ 3584
#define NH_  16
#define NKV_ 8
#define HD_  256
#define WIN_ 1024
#define QKVN 8192   // (NH+2*NKV)*HD
#define ON_  4096   // NH*HD

// ---- CDNA5 async global->LDS path (probe-guarded) ----
#if defined(__has_builtin)
#if __has_builtin(__builtin_amdgcn_global_load_async_to_lds_b128)
#define ASYNC_LDS 1
#endif
#endif

__device__ __forceinline__ void async_copy16(const void* g, void* l) {
#if defined(ASYNC_LDS)
  __builtin_amdgcn_global_load_async_to_lds_b128(
      (__attribute__((address_space(1))) v4i_*)g,
      (__attribute__((address_space(3))) v4i_*)l, 0, 0);
#else
  *(float4*)l = *(const float4*)g;
#endif
}

__device__ __forceinline__ void pipe_sync() {
#if defined(ASYNC_LDS)
#if __has_builtin(__builtin_amdgcn_s_wait_asynccnt)
  __builtin_amdgcn_s_wait_asynccnt(0);
#else
  asm volatile("s_wait_asynccnt 0x0" ::: "memory");
#endif
#endif
  __syncthreads();
}

__device__ __forceinline__ v8f wmma_bf16(v16bf a, v16bf b, v8f c) {
  return __builtin_amdgcn_wmma_f32_16x16x32_bf16(
      false, a, false, b, (short)0, c, false, false);
}

union pk4 { __bf16 h[4]; float2 v; };
union pk8 { __bf16 h[8]; float4 v; };

// ---------------------------------------------------------------------------
// Tiled GEMM: C[M,N] = A[M,K] * B[K,N].  Block tile 128x128, K-step 32,
// double-buffered LDS. 256 threads = 8 waves (2M x 4N); each wave owns a
// 64x32 tile = 4x2 grid of 16x16x32 WMMA fragments.
// ---------------------------------------------------------------------------
template <bool A_BF16, bool C_BF16>
__global__ __launch_bounds__(256) void gemm_wmma_kernel(
    const void* __restrict__ Avoid, const float* __restrict__ B,
    void* __restrict__ Cvoid, int M, int N, int K) {
  constexpr int BM = 128, BN = 128, BK = 32;
  __shared__ __align__(16) __bf16 sA[2][BM][BK];   // [m][k]
  __shared__ __align__(16) __bf16 sBT[2][BN][BK];  // transposed [n][k]

  const int tid  = threadIdx.x;
  const int lane = tid & 31;
  const int wave = tid >> 5;
  const int hi16 = lane >> 4;
  const int l15  = lane & 15;
  const int wm   = (wave & 1) * 64;
  const int wn   = (wave >> 1) * 32;
  const int bm   = blockIdx.x * BM;
  const int bn   = blockIdx.y * BN;

  const float*  Af = (const float*)Avoid;
  const __bf16* Ab = (const __bf16*)Avoid;

  const v8f vz = {0.f, 0.f, 0.f, 0.f, 0.f, 0.f, 0.f, 0.f};
  v8f acc[4][2];
#pragma unroll
  for (int i = 0; i < 4; ++i)
#pragma unroll
    for (int j = 0; j < 2; ++j) acc[i][j] = vz;

  auto stage = [&](int buf, int k0) {
    // ---- A tile (128x32) ----
    if (A_BF16) {
#pragma unroll
      for (int i = 0; i < 2; ++i) {        // 512 chunks of 8 bf16
        int s = tid + i * 256;
        int row = s >> 2;
        int c8 = (s & 3) << 3;
        async_copy16(Ab + (size_t)(bm + row) * K + k0 + c8, &sA[buf][row][c8]);
      }
    } else {
#pragma unroll
      for (int i = 0; i < 4; ++i) {        // 1024 float4 slots
        int s = tid + i * 256;
        int row = s >> 3;
        int c4 = (s & 7) << 2;
        float4 a4 = *(const float4*)(Af + (size_t)(bm + row) * K + k0 + c4);
        pk4 u;
        u.h[0] = (__bf16)a4.x; u.h[1] = (__bf16)a4.y;
        u.h[2] = (__bf16)a4.z; u.h[3] = (__bf16)a4.w;
        *(float2*)(&sA[buf][row][c4]) = u.v;   // packed b64 store
      }
    }
    // ---- B tile (32x128) -> transposed [n][k], packed b64 stores ----
#pragma unroll
    for (int i = 0; i < 4; ++i) {
      int s = tid + i * 256;
      int n = s & 127;
      int kq = (s >> 7) << 2;
      pk4 u;
#pragma unroll
      for (int j = 0; j < 4; ++j)
        u.h[j] = (__bf16)B[(size_t)(k0 + kq + j) * N + bn + n];
      *(float2*)(&sBT[buf][n][kq]) = u.v;
    }
  };

  stage(0, 0);
  pipe_sync();
  int cur = 0;

  for (int k0 = 0; k0 < K; k0 += BK) {
    if (k0 + BK < K) {
      stage(cur ^ 1, k0 + BK);
      if (k0 + 2 * BK < K)  // global_prefetch_b8 hint two tiles ahead
        __builtin_prefetch(B + (size_t)(k0 + 2 * BK) * N + bn + (tid << 1), 0, 3);
    }

    v16bf afr[4];
#pragma unroll
    for (int i = 0; i < 4; ++i) {
      const __bf16* base = &sA[cur][wm + i * 16 + l15][0];
      int koff = hi16 * 16;
#pragma unroll
      for (int j = 0; j < 16; ++j)
        afr[i][j] = base[j + koff];
    }
    v16bf bfr[2];
#pragma unroll
    for (int j = 0; j < 2; ++j) {
      const __bf16* base = &sBT[cur][wn + j * 16 + l15][hi16 * 16];
#pragma unroll
      for (int t = 0; t < 16; ++t) bfr[j][t] = base[t];
    }
#pragma unroll
    for (int i = 0; i < 4; ++i)
#pragma unroll
      for (int j = 0; j < 2; ++j)
        acc[i][j] = wmma_bf16(afr[i], bfr[j], acc[i][j]);

    pipe_sync();
    cur ^= 1;
  }

  // ---- epilogue: C frag element (M = v + 8*hi16, N = l15) ----
#pragma unroll
  for (int i = 0; i < 4; ++i) {
#pragma unroll
    for (int j = 0; j < 2; ++j) {
#pragma unroll
      for (int v = 0; v < 8; ++v) {
        int row = bm + wm + i * 16 + v + hi16 * 8;
        int col = bn + wn + j * 16 + l15;
        if (C_BF16)
          ((__bf16*)Cvoid)[(size_t)row * N + col] = (__bf16)acc[i][j][v];
        else
          ((float*)Cvoid)[(size_t)row * N + col] = acc[i][j][v];
      }
    }
  }
}

// ---------------------------------------------------------------------------
// RoPE: reads bf16 qkv, writes head-major bf16 Q (x SCALING/SOFTCAP) and K.
// ---------------------------------------------------------------------------
__global__ __launch_bounds__(256) void rope_kernel(
    const __bf16* __restrict__ qkv, const float* __restrict__ fcos,
    const float* __restrict__ fsin, __bf16* __restrict__ Qb,
    __bf16* __restrict__ Kb) {
  int idx = blockIdx.x * 256 + threadIdx.x;
  const int QP = S_ * NH_ * 128;
  const int KP = S_ * NKV_ * 128;
  if (idx < QP) {
    int p = idx & 127;
    int h = (idx >> 7) & 15;
    int s = idx >> 11;
    size_t base = (size_t)s * QKVN + h * HD_;
    float x1 = (float)qkv[base + p];
    float x2 = (float)qkv[base + 128 + p];
    float c = fcos[s * 128 + p], sn = fsin[s * 128 + p];
    const float f = 0.0625f / 50.0f;  // SCALING / SOFTCAP folded into Q
    size_t o = ((size_t)h * S_ + s) * HD_;
    Qb[o + p]       = (__bf16)((x1 * c - x2 * sn) * f);
    Qb[o + 128 + p] = (__bf16)((x1 * sn + x2 * c) * f);
  } else if (idx < QP + KP) {
    int t = idx - QP;
    int p = t & 127;
    int h = (t >> 7) & 7;
    int s = t >> 10;
    size_t base = (size_t)s * QKVN + NH_ * HD_ + h * HD_;
    float x1 = (float)qkv[base + p];
    float x2 = (float)qkv[base + 128 + p];
    float c = fcos[s * 128 + p], sn = fsin[s * 128 + p];
    size_t o = ((size_t)h * S_ + s) * HD_;
    Kb[o + p]       = (__bf16)(x1 * c - x2 * sn);
    Kb[o + 128 + p] = (__bf16)(x1 * sn + x2 * c);
  }
}

// ---------------------------------------------------------------------------
// Flash attention, sliding window + causal + tanh softcap.
// Block = (128 q rows, 1 head); 8 waves, 16 q rows/wave. Q in registers
// (8 A-frags). K tile [32][256] (async-staged) and V^T [256][32] double-
// buffered in LDS. P re-layout C->A through wave-private LDS. Fragment
// loads are software-pipelined one ahead of each WMMA.
// ---------------------------------------------------------------------------
__global__ __launch_bounds__(256) void attn_kernel(
    const __bf16* __restrict__ Qb, const __bf16* __restrict__ Kb,
    const __bf16* __restrict__ qkv, __bf16* __restrict__ Ob) {
  __shared__ __align__(16) __bf16 sK[2][32][HD_];   // 32 KB
  __shared__ __align__(16) __bf16 sVT[2][HD_][32];  // 32 KB
  __shared__ __align__(16) __bf16 sP[8][16][32];    // 8 KB, wave-private

  const int tid  = threadIdx.x;
  const int lane = tid & 31;
  const int wave = tid >> 5;
  const int hi16 = lane >> 4;
  const int l15  = lane & 15;
  const int head = blockIdx.y;
  const int kvh  = head >> 1;          // GQA group of 2
  const int qblk = blockIdx.x * 128;
  const int qrow0 = qblk + wave * 16;

  // ---- Q fragments (16 rows x 256 = 8 A-frags) ----
  v16bf qf[8];
  const __bf16* Qp = Qb + ((size_t)head * S_ + qrow0 + l15) * HD_;
#pragma unroll
  for (int f = 0; f < 8; ++f) {
    int koff = hi16 * 16 + f * 32;
#pragma unroll
    for (int j = 0; j < 16; ++j)
      qf[f][j] = Qp[j + koff];
  }

  const v8f vz = {0.f, 0.f, 0.f, 0.f, 0.f, 0.f, 0.f, 0.f};
  v8f o[16];
#pragma unroll
  for (int f = 0; f < 16; ++f) o[f] = vz;
  float mrow[8], lrow[8];
#pragma unroll
  for (int v = 0; v < 8; ++v) { mrow[v] = -1e30f; lrow[v] = 0.f; }

  int lo = qblk - (WIN_ - 1);
  if (lo < 0) lo = 0;
  lo &= ~31;
  const int hi = qblk + 127;

  auto stage = [&](int buf, int kv0) {
    // K rows [kv0, kv0+31] -> async b128 straight into LDS
#pragma unroll
    for (int i = 0; i < 4; ++i) {
      int s = tid + i * 256;
      int r = s >> 5;
      int c8 = (s & 31) << 3;
      async_copy16(Kb + ((size_t)kvh * S_ + kv0 + r) * HD_ + c8,
                   &sK[buf][r][c8]);
    }
    // V (inside qkv, col 6144) transposed; packed b128 stores
#pragma unroll
    for (int i = 0; i < 32; i += 8) {
      pk8 u;
#pragma unroll
      for (int j = 0; j < 8; ++j)
        u.h[j] = qkv[(size_t)(kv0 + i + j) * QKVN + (NH_ + NKV_) * HD_ +
                     kvh * HD_ + tid];
      *(float4*)(&sVT[buf][tid][i]) = u.v;
    }
  };

  stage(0, lo);
  pipe_sync();
  int cur = 0;

  for (int kv0 = lo; kv0 <= hi; kv0 += 32) {
    if (kv0 + 32 <= hi) stage(cur ^ 1, kv0 + 32);

    // ---- S = Q @ K^T : two 16x16 C-frags, frag loads pipelined ----
    v8f s0 = vz, s1 = vz;
    v16bf b0c, b1c, b0n, b1n;
    {
      const __bf16* k0p = &sK[cur][l15][hi16 * 16];
      const __bf16* k1p = &sK[cur][16 + l15][hi16 * 16];
#pragma unroll
      for (int t = 0; t < 16; ++t) { b0c[t] = k0p[t]; b1c[t] = k1p[t]; }
    }
#pragma unroll
    for (int f = 0; f < 8; ++f) {
      if (f < 7) {
        const __bf16* k0p = &sK[cur][l15][(f + 1) * 32 + hi16 * 16];
        const __bf16* k1p = &sK[cur][16 + l15][(f + 1) * 32 + hi16 * 16];
#pragma unroll
        for (int t = 0; t < 16; ++t) { b0n[t] = k0p[t]; b1n[t] = k1p[t]; }
      }
      s0 = wmma_bf16(qf[f], b0c, s0);
      s1 = wmma_bf16(qf[f], b1c, s1);
      if (f < 7) { b0c = b0n; b1c = b1n; }
    }

    // ---- softcap + band/causal mask + online softmax ----
    float scale[8];
#pragma unroll
    for (int v = 0; v < 8; ++v) {
      int qg = qrow0 + v + hi16 * 8;
      int kg0 = kv0 + l15;
      int kg1 = kg0 + 16;
      float x0 = s0[v], x1 = s1[v];
      bool ok0 = (kg0 <= qg) && (qg - kg0 < WIN_);
      bool ok1 = (kg1 <= qg) && (qg - kg1 < WIN_);
      x0 = ok0 ? tanhf(x0) * 50.f : -1e30f;
      x1 = ok1 ? tanhf(x1) * 50.f : -1e30f;
      float rmax = fmaxf(x0, x1);
#pragma unroll
      for (int off = 8; off >= 1; off >>= 1)
        rmax = fmaxf(rmax, __shfl_xor(rmax, off, 32));
      float mnew = fmaxf(mrow[v], rmax);
      float p0 = (x0 > -1e29f) ? __expf(x0 - mnew) : 0.f;
      float p1 = (x1 > -1e29f) ? __expf(x1 - mnew) : 0.f;
      float ps = p0 + p1;
#pragma unroll
      for (int off = 8; off >= 1; off >>= 1) ps += __shfl_xor(ps, off, 32);
      scale[v] = __expf(mrow[v] - mnew);
      lrow[v] = lrow[v] * scale[v] + ps;
      mrow[v] = mnew;
      s0[v] = p0;
      s1[v] = p1;
    }
#pragma unroll
    for (int f = 0; f < 16; ++f)
#pragma unroll
      for (int v = 0; v < 8; ++v) o[f][v] *= scale[v];

    // ---- P: C-layout -> A-layout via wave-private LDS ----
#pragma unroll
    for (int v = 0; v < 8; ++v) {
      sP[wave][v + hi16 * 8][l15]      = (__bf16)s0[v];
      sP[wave][v + hi16 * 8][16 + l15] = (__bf16)s1[v];
    }
    v16bf pf;
    {
      const __bf16* pp = &sP[wave][l15][hi16 * 16];
#pragma unroll
      for (int j = 0; j < 16; ++j) pf[j] = pp[j];
    }

    // ---- O += P @ V : 16 hd-frags, frag loads pipelined ----
    v16bf vbc, vbn;
    {
      const __bf16* vp = &sVT[cur][l15][hi16 * 16];
#pragma unroll
      for (int t = 0; t < 16; ++t) vbc[t] = vp[t];
    }
#pragma unroll
    for (int f = 0; f < 16; ++f) {
      if (f < 15) {
        const __bf16* vp = &sVT[cur][(f + 1) * 16 + l15][hi16 * 16];
#pragma unroll
        for (int t = 0; t < 16; ++t) vbn[t] = vp[t];
      }
      o[f] = wmma_bf16(pf, vbc, o[f]);
      if (f < 15) vbc = vbn;
    }

    pipe_sync();
    cur ^= 1;
  }

  // ---- normalize and store O as bf16 [S][NH*HD] ----
  float inv[8];
#pragma unroll
  for (int v = 0; v < 8; ++v) inv[v] = 1.f / fmaxf(lrow[v], 1e-30f);
#pragma unroll
  for (int f = 0; f < 16; ++f) {
#pragma unroll
    for (int v = 0; v < 8; ++v) {
      int row = qrow0 + v + hi16 * 8;
      int col = head * HD_ + f * 16 + l15;
      Ob[(size_t)row * ON_ + col] = (__bf16)(o[f][v] * inv[v]);
    }
  }
}

// ---------------------------------------------------------------------------
extern "C" void kernel_launch(void* const* d_in, const int* in_sizes, int n_in,
                              void* d_out, int out_size, void* d_ws,
                              size_t ws_size, hipStream_t stream) {
  (void)in_sizes; (void)n_in; (void)out_size; (void)ws_size;
  const float* hidden = (const float*)d_in[0];  // [2048][3584]
  const float* Wqkv   = (const float*)d_in[1];  // [3584][8192]
  const float* Wo     = (const float*)d_in[2];  // [4096][3584]
  const float* fcos   = (const float*)d_in[3];  // [2048][128]
  const float* fsin   = (const float*)d_in[4];  // [2048][128]
  // d_in[5] (mask) is implemented analytically in the attention kernel.

  char* ws = (char*)d_ws;
  __bf16* qkv = (__bf16*)(ws);                  // 2048*8192*2 = 33,554,432 B
  __bf16* Qb  = (__bf16*)(ws + 33554432);       // 16*2048*256*2 = 16,777,216 B
  __bf16* Kb  = (__bf16*)(ws + 50331648);       //  8*2048*256*2 =  8,388,608 B
  __bf16* Ob  = (__bf16*)(ws + 58720256);       // 2048*4096*2   = 16,777,216 B

  dim3 g1(S_ / 128, QKVN / 128);
  gemm_wmma_kernel<false, true><<<g1, 256, 0, stream>>>(
      (const void*)hidden, Wqkv, (void*)qkv, S_, QKVN, HID_);

  int total = S_ * NH_ * 128 + S_ * NKV_ * 128;  // 6,291,456
  rope_kernel<<<total / 256, 256, 0, stream>>>(qkv, fcos, fsin, Qb, Kb);

  dim3 ga(S_ / 128, NH_);
  attn_kernel<<<ga, 256, 0, stream>>>(Qb, Kb, qkv, Ob);

  dim3 g2(S_ / 128, HID_ / 128);
  gemm_wmma_kernel<true, false><<<g2, 256, 0, stream>>>(
      (const void*)Ob, Wo, d_out, S_, HID_, ON_);
}
